// AttentiveFPModel_11733850653137
// MI455X (gfx1250) — matrix operand order repaired and checked by reading, not compiled, not run on hardware
//
#include <hip/hip_runtime.h>

#define WPB 8   // waves per 256-thread block (wave32)
#define TPW 4   // tiles (16 rows each) per wave in edge/readout kernels
#define WPBG 4  // waves per block in the GRU kernel (128 threads)

typedef __attribute__((ext_vector_type(16))) _Float16 v16h;
typedef __attribute__((ext_vector_type(8)))  float    v8f;

__device__ __forceinline__ float lrelu01(float x){ return x >= 0.f ? x : 0.01f * x; }
__device__ __forceinline__ float sigm(float x){ return 1.f / (1.f + expf(-x)); }
__device__ __forceinline__ long  clampl(long v, long hi){ return v < hi ? v : hi; }

// monotonic float atomic-max via int/uint atomics
__device__ __forceinline__ void atomicMaxF(float* a, float v){
  if (v >= 0.f) atomicMax((int*)a, __float_as_int(v));
  else          atomicMin((unsigned int*)a, __float_as_uint(v));
}

__device__ __forceinline__ v8f wmma16(v16h a, v16h b, v8f c){
  // D = A(16x32 f16) * B(32x16 f16) + C(16x16 f32)
  return __builtin_amdgcn_wmma_f32_16x16x32_f16(false, a, false, b, (short)0, c, false, false);
}

// A fragment (16x32 f16) for this lane from a contiguous f16 row (>= kbase+32 elems).
// lane<16: row m=lane, K = kbase+0..15; lane>=16: K = kbase+16..31
__device__ __forceinline__ v16h a_frag(const _Float16* rowp, int kbase, int hf){
  v16h a;
  const _Float16* p = rowp + kbase + hf * 16;
  #pragma unroll
  for (int i = 0; i < 16; ++i) a[i] = p[i];
  return a;
}

// B fragment (32x16 f16): B[k][n] = W[nbase+n][kbase+k], W row-major, leading dim ld.
// lane column n = lane&15; lanes<16 hold K 0..15, lanes>=16 hold K 16..31 (contiguous in W row).
__device__ __forceinline__ v16h b_frag(const _Float16* W, int ld, int nbase, int kbase, int lid, int hf){
  v16h b;
  const _Float16* p = W + (size_t)(nbase + lid) * ld + kbase + hf * 16;
  #pragma unroll
  for (int i = 0; i < 16; ++i) b[i] = p[i];
  return b;
}

// xj tile for 16 edges with pre-hoisted B fragments.
// acc[j]: C-layout tile cols [16j..16j+15], rows m = r + 8*hf.
// xj = leaky_relu( [x[src] | edge_attr] @ gc_lin1_w^T ), edge_attr column folded into C-init.
__device__ __forceinline__ void compute_xj_pre(const _Float16* xh, const float* edge_attr,
    const int* src, const v16h b1[4][2], const float wlast[4],
    long e0, long E, int lid, int hf, v8f acc[4])
{
  const _Float16* arow = xh + (size_t)src[clampl(e0 + lid, E - 1)] * 64;
  v16h a0 = a_frag(arow, 0, hf);
  v16h a1 = a_frag(arow, 32, hf);
  #pragma unroll
  for (int j = 0; j < 4; ++j){
    v8f c;
    #pragma unroll
    for (int r = 0; r < 8; ++r){
      long e = clampl(e0 + r + hf * 8, E - 1);
      c[r] = edge_attr[e] * wlast[j];
    }
    c = wmma16(a0, b1[j][0], c);
    c = wmma16(a1, b1[j][1], c);
    #pragma unroll
    for (int r = 0; r < 8; ++r) c[r] = lrelu01(c[r]);
    acc[j] = c;
  }
}

// ---------------- prep kernels ----------------

__global__ void k_init(float* hacc, float* amax, float* ssum, float* gpool,
                       float* g_h2, float* g_sum, float* g_amax, long Nn){
  long i = (long)blockIdx.x * blockDim.x + threadIdx.x;
  if (i < Nn * 64) hacc[i] = 0.f;
  if (i < Nn){ amax[i] = -INFINITY; ssum[i] = 0.f; }
  if (i < 64){ gpool[i] = 0.f; g_h2[i] = 0.f; }
  if (i == 0){ *g_sum = 0.f; *g_amax = -INFINITY; }
}

__global__ void k_conv_w(const float* gc1, const float* gc2, const float* wih,
                         const float* whh, const float* mcl,
                         _Float16* w1h, _Float16* w2h, _Float16* wihh,
                         _Float16* whhh, _Float16* mclh){
  int i = blockIdx.x * blockDim.x + threadIdx.x;
  if (i < 64 * 65) w1h[i] = (_Float16)gc1[i];
  if (i < 64 * 64){ w2h[i] = (_Float16)gc2[i]; mclh[i] = (_Float16)mcl[i]; }
  if (i < 192 * 64){ wihh[i] = (_Float16)wih[i]; whhh[i] = (_Float16)whh[i]; }
}

// x = leaky_relu(node_attr * lin1_w + lin1_b); f16 copy; dst-side attention dot.
// One wave per node: lane owns cols {lane, lane+32} (coalesced), shfl reduction for the dot.
__global__ __launch_bounds__(256) void k_node_prep(const float* node_attr, const float* lin1_w,
    const float* lin1_b, const float* att_r, float* x, _Float16* xh, float* dstdot, long Nn){
  int tid = threadIdx.x, lane = tid & 31, wid = tid >> 5;
  long n = (long)blockIdx.x * WPB + wid;
  if (n >= Nn) return;
  float a = node_attr[n];
  float dd = 0.f;
  #pragma unroll
  for (int s = 0; s < 2; ++s){
    int c = lane + s * 32;
    float v = lrelu01(a * lin1_w[c] + lin1_b[c]);
    x[n * 64 + c] = v;
    xh[n * 64 + c] = (_Float16)v;
    dd += v * att_r[c];
  }
  dd += __shfl_xor(dd, 1, 32);  dd += __shfl_xor(dd, 2, 32);
  dd += __shfl_xor(dd, 4, 32);  dd += __shfl_xor(dd, 8, 32);
  dd += __shfl_xor(dd, 16, 32);
  if (lane == 0) dstdot[n] = dd;
}

// ---------------- edge passes ----------------

// pass1: xj GEMM (WMMA), attention logits, per-dst segment max.  4 tiles per wave,
// B fragments / att vectors hoisted into registers.
__global__ __launch_bounds__(256) void k_edge1(const _Float16* xh, const float* edge_attr,
    const int* src, const int* dst, const _Float16* w1h, const float* gc1,
    const float* att_l, const float* dstdot, float* alpha, float* amax, long E)
{
  int tid = threadIdx.x, lane = tid & 31, wid = tid >> 5;
  int lid = lane & 15, hf = lane >> 4;
  long etiles = (E + 15) / 16;
  long tile0 = ((long)blockIdx.x * WPB + wid) * TPW;
  if (tile0 >= etiles) return;

  v16h b1[4][2]; float wlast[4], al[4];
  #pragma unroll
  for (int j = 0; j < 4; ++j){
    b1[j][0] = b_frag(w1h, 65, j * 16, 0,  lid, hf);
    b1[j][1] = b_frag(w1h, 65, j * 16, 32, lid, hf);
    wlast[j] = gc1[(j * 16 + lid) * 65 + 64];   // edge_attr column, exact f32
    al[j]    = att_l[j * 16 + lid];
  }

  for (int t = 0; t < TPW; ++t){
    long tile = tile0 + t;
    if (tile >= etiles) break;                   // wave-uniform
    long e0 = tile * 16;
    v8f acc[4];
    compute_xj_pre(xh, edge_attr, src, b1, wlast, e0, E, lid, hf, acc);
    #pragma unroll
    for (int r = 0; r < 8; ++r){
      float p = 0.f;
      #pragma unroll
      for (int j = 0; j < 4; ++j) p += acc[j][r] * al[j];
      p += __shfl_xor(p, 1, 32); p += __shfl_xor(p, 2, 32);
      p += __shfl_xor(p, 4, 32); p += __shfl_xor(p, 8, 32);
      if (lid == 0){
        long e = e0 + r + hf * 8;
        if (e < E){
          int d = dst[e];
          float ar = lrelu01(p + dstdot[d]);
          alpha[e] = ar;
          atomicMaxF(&amax[d], ar);
        }
      }
    }
  }
}

// pass2: exp(alpha - amax[dst]), per-dst sum
__global__ void k_edge2(const int* dst, float* alpha, const float* amax, float* ssum, long E){
  long e = (long)blockIdx.x * blockDim.x + threadIdx.x;
  if (e >= E) return;
  int d = dst[e];
  float ex = expf(alpha[e] - amax[d]);
  alpha[e] = ex;
  atomicAdd(&ssum[d], ex);
}

// pass3: recompute xj (WMMA), wave-local LDS transpose, msg = xj @ W2^T (WMMA),
// scale by normalized alpha, scatter-add.  4 tiles per wave, all B fragments hoisted.
__global__ __launch_bounds__(256) void k_edge3(const _Float16* xh, const float* edge_attr,
    const int* src, const int* dst, const _Float16* w1h, const float* gc1,
    const _Float16* w2h, const float* alpha, const float* ssum, float* hacc, long E)
{
  __shared__ _Float16 s_xj[WPB * 16 * 64];
  int tid = threadIdx.x, lane = tid & 31, wid = tid >> 5;
  int lid = lane & 15, hf = lane >> 4;
  long etiles = (E + 15) / 16;
  long tile0 = ((long)blockIdx.x * WPB + wid) * TPW;
  if (tile0 >= etiles) return;

  v16h b1[4][2], b2[4][2]; float wlast[4];
  #pragma unroll
  for (int j = 0; j < 4; ++j){
    b1[j][0] = b_frag(w1h, 65, j * 16, 0,  lid, hf);
    b1[j][1] = b_frag(w1h, 65, j * 16, 32, lid, hf);
    b2[j][0] = b_frag(w2h, 64, j * 16, 0,  lid, hf);
    b2[j][1] = b_frag(w2h, 64, j * 16, 32, lid, hf);
    wlast[j] = gc1[(j * 16 + lid) * 65 + 64];
  }
  _Float16* myxj = s_xj + wid * 1024;

  for (int t = 0; t < TPW; ++t){
    long tile = tile0 + t;
    if (tile >= etiles) break;
    long e0 = tile * 16;

    v8f acc[4];
    compute_xj_pre(xh, edge_attr, src, b1, wlast, e0, E, lid, hf, acc);

    #pragma unroll
    for (int j = 0; j < 4; ++j)
      #pragma unroll
      for (int r = 0; r < 8; ++r)
        myxj[(r + hf * 8) * 64 + j * 16 + lid] = (_Float16)acc[j][r];

    // wave-local C-layout -> A-layout transpose through LDS (same wave, DScnt-ordered)
    v16h a0 = a_frag(myxj + lid * 64, 0, hf);
    v16h a1 = a_frag(myxj + lid * 64, 32, hf);

    float an[8];
    #pragma unroll
    for (int r = 0; r < 8; ++r){
      long e = clampl(e0 + r + hf * 8, E - 1);
      an[r] = alpha[e] / (ssum[dst[e]] + 1e-16f);
    }

    #pragma unroll
    for (int j = 0; j < 4; ++j){
      v8f c;
      #pragma unroll
      for (int r = 0; r < 8; ++r) c[r] = 0.f;
      c = wmma16(a0, b2[j][0], c);
      c = wmma16(a1, b2[j][1], c);
      #pragma unroll
      for (int r = 0; r < 8; ++r){
        long e = e0 + r + hf * 8;
        if (e < E)
          atomicAdd(hacc + (size_t)dst[e] * 64 + j * 16 + lid, c[r] * an[r]);
      }
    }
  }
}

// ---------------- node GRU + pool (LDS-staged weights) ----------------

__global__ __launch_bounds__(128) void k_node_gru(
    const float* hacc, const float* gc_bias, const float* x, const _Float16* xh,
    const _Float16* wih, const _Float16* whh, const float* bih, const float* bhh,
    float* x2, _Float16* x2h, float* gpool, long Nn)
{
  __shared__ __align__(16) _Float16 s_wi[192 * 64];
  __shared__ __align__(16) _Float16 s_wh[192 * 64];
  __shared__ _Float16 s_h[WPBG * 16 * 64];
  __shared__ float spool[64];
  int tid = threadIdx.x;
  if (tid < 64) spool[tid] = 0.f;
  {
    // stage both GRU weight matrices into LDS (16B granules, 12288 halves each)
    const uint4* si = (const uint4*)wih;
    const uint4* sh = (const uint4*)whh;
    uint4* di = (uint4*)s_wi;
    uint4* dh = (uint4*)s_wh;
    for (int i = tid; i < 1536; i += 128){ di[i] = si[i]; dh[i] = sh[i]; }
  }
  __syncthreads();

  int lane = tid & 31, wid = tid >> 5, lid = lane & 15, hf = lane >> 4;
  long tile = (long)blockIdx.x * WPBG + wid;
  if (tile < (Nn + 15) / 16){
    long n0 = tile * 16;
    _Float16* myh = s_h + wid * 1024;
    {
      long node = clampl(n0 + lid, Nn - 1);
      const float* hrow = hacc + node * 64;
      #pragma unroll 4
      for (int c = 0; c < 32; ++c){
        int col = hf * 32 + c;
        float v = hrow[col] + gc_bias[col];
        v = v > 0.f ? v : (expf(v) - 1.f);       // ELU
        myh[lid * 64 + col] = (_Float16)v;
      }
    }
    const _Float16* xrow = xh + (size_t)clampl(n0 + lid, Nn - 1) * 64;
    v16h ah0 = a_frag(s_h + wid * 1024 + lid * 64, 0, hf);
    v16h ah1 = a_frag(s_h + wid * 1024 + lid * 64, 32, hf);
    v16h ax0 = a_frag(xrow, 0, hf);
    v16h ax1 = a_frag(xrow, 32, hf);
    #pragma unroll
    for (int j = 0; j < 4; ++j){
      int col = j * 16 + lid;
      v8f ir, iz, in2, hr, hz, hn;
      #pragma unroll
      for (int r = 0; r < 8; ++r){
        ir[r] = bih[col]; iz[r] = bih[64 + col]; in2[r] = bih[128 + col];
        hr[r] = bhh[col]; hz[r] = bhh[64 + col]; hn[r]  = bhh[128 + col];
      }
      ir  = wmma16(ah0, b_frag(s_wi, 64,       j * 16, 0,  lid, hf), ir);
      ir  = wmma16(ah1, b_frag(s_wi, 64,       j * 16, 32, lid, hf), ir);
      iz  = wmma16(ah0, b_frag(s_wi, 64,  64 + j * 16, 0,  lid, hf), iz);
      iz  = wmma16(ah1, b_frag(s_wi, 64,  64 + j * 16, 32, lid, hf), iz);
      in2 = wmma16(ah0, b_frag(s_wi, 64, 128 + j * 16, 0,  lid, hf), in2);
      in2 = wmma16(ah1, b_frag(s_wi, 64, 128 + j * 16, 32, lid, hf), in2);
      hr  = wmma16(ax0, b_frag(s_wh, 64,       j * 16, 0,  lid, hf), hr);
      hr  = wmma16(ax1, b_frag(s_wh, 64,       j * 16, 32, lid, hf), hr);
      hz  = wmma16(ax0, b_frag(s_wh, 64,  64 + j * 16, 0,  lid, hf), hz);
      hz  = wmma16(ax1, b_frag(s_wh, 64,  64 + j * 16, 32, lid, hf), hz);
      hn  = wmma16(ax0, b_frag(s_wh, 64, 128 + j * 16, 0,  lid, hf), hn);
      hn  = wmma16(ax1, b_frag(s_wh, 64, 128 + j * 16, 32, lid, hf), hn);
      float pp = 0.f;
      #pragma unroll
      for (int r = 0; r < 8; ++r){
        long nidx = n0 + r + hf * 8;
        long node = clampl(nidx, Nn - 1);
        float rg = sigm(ir[r] + hr[r]);
        float zg = sigm(iz[r] + hz[r]);
        float ng = tanhf(in2[r] + rg * hn[r]);
        float v = (1.f - zg) * ng + zg * x[node * 64 + col];
        v = v > 0.f ? v : 0.f;                   // ReLU
        if (nidx < Nn){
          x2[node * 64 + col]  = v;
          x2h[node * 64 + col] = (_Float16)v;
          pp += v;
        }
      }
      atomicAdd(&spool[col], pp);
    }
  }
  __syncthreads();
  if (tid < 64) atomicAdd(&gpool[tid], spool[tid]);
}

// ---------------- readout ----------------

// out = relu(pool); xd = out @ mc_lin^T; c0 = xd . att_dst   (1 block, 64 threads)
__global__ void k_pool_fin(const float* gpool, const float* mclin, const float* att_dst,
                           float* outv, float* c0){
  __shared__ float s_out[64], s_red[64];
  int c = threadIdx.x;
  float o = gpool[c]; o = o > 0.f ? o : 0.f;
  outv[c] = o; s_out[c] = o;
  __syncthreads();
  float xd = 0.f;
  for (int j = 0; j < 64; ++j) xd += mclin[c * 64 + j] * s_out[j];
  s_red[c] = xd * att_dst[c];
  __syncthreads();
  if (c == 0){ float s = 0.f; for (int i = 0; i < 64; ++i) s += s_red[i]; *c0 = s; }
}

// xs = x2 @ mc_lin^T (WMMA), a_raw = lrelu(xs.att_src + c0), global segment max.
// 4 tiles per wave, mc_lin fragments hoisted.
__global__ __launch_bounds__(256) void k_readout_xs(const _Float16* x2h, const _Float16* mclh,
    const float* att_src, const float* c0, float* xs, float* araw, float* g_amax, long Nn)
{
  int tid = threadIdx.x, lane = tid & 31, wid = tid >> 5;
  int lid = lane & 15, hf = lane >> 4;
  long ntiles = (Nn + 15) / 16;
  long tile0 = ((long)blockIdx.x * WPB + wid) * TPW;
  if (tile0 >= ntiles) return;

  v16h bm[4][2]; float as[4];
  #pragma unroll
  for (int j = 0; j < 4; ++j){
    bm[j][0] = b_frag(mclh, 64, j * 16, 0,  lid, hf);
    bm[j][1] = b_frag(mclh, 64, j * 16, 32, lid, hf);
    as[j]    = att_src[j * 16 + lid];
  }
  float c0v = *c0;

  for (int t = 0; t < TPW; ++t){
    long tile = tile0 + t;
    if (tile >= ntiles) break;
    long n0 = tile * 16;

    const _Float16* arow = x2h + (size_t)clampl(n0 + lid, Nn - 1) * 64;
    v16h a0 = a_frag(arow, 0, hf);
    v16h a1 = a_frag(arow, 32, hf);
    v8f acc[4];
    #pragma unroll
    for (int j = 0; j < 4; ++j){
      v8f c;
      #pragma unroll
      for (int r = 0; r < 8; ++r) c[r] = 0.f;
      c = wmma16(a0, bm[j][0], c);
      c = wmma16(a1, bm[j][1], c);
      acc[j] = c;
      #pragma unroll
      for (int r = 0; r < 8; ++r){
        long nidx = n0 + r + hf * 8;
        if (nidx < Nn) xs[(size_t)nidx * 64 + j * 16 + lid] = c[r];
      }
    }
    float lmax = -INFINITY;
    #pragma unroll
    for (int r = 0; r < 8; ++r){
      float p = 0.f;
      #pragma unroll
      for (int j = 0; j < 4; ++j) p += acc[j][r] * as[j];
      p += __shfl_xor(p, 1, 32); p += __shfl_xor(p, 2, 32);
      p += __shfl_xor(p, 4, 32); p += __shfl_xor(p, 8, 32);
      if (lid == 0){
        long nidx = n0 + r + hf * 8;
        if (nidx < Nn){
          float ar = lrelu01(p + c0v);
          araw[nidx] = ar;
          lmax = lmax > ar ? lmax : ar;
        }
      }
    }
    if (lid == 0) atomicMaxF(g_amax, lmax);
  }
}

// single-segment weighted sum: g_sum += exp, g_h2 += xs * exp  (block-local LDS then 1 atomic)
__global__ void k_readout_sum(const float* xs, const float* araw, const float* g_amax,
                              float* g_sum, float* g_h2, long Nn){
  __shared__ float svec[64];
  __shared__ float s_se;
  int tid = threadIdx.x;
  if (tid < 64) svec[tid] = 0.f;
  if (tid == 0) s_se = 0.f;
  __syncthreads();
  long i = (long)blockIdx.x * blockDim.x + tid;
  if (i < Nn){
    float e = expf(araw[i] - *g_amax);
    atomicAdd(&s_se, e);
    const float* row = xs + (size_t)i * 64;
    for (int k = 0; k < 64; ++k){
      int c = (tid + k) & 63;               // stagger to spread LDS bank/address conflicts
      atomicAdd(&svec[c], row[c] * e);
    }
  }
  __syncthreads();
  if (tid < 64) atomicAdd(&g_h2[tid], svec[tid]);
  if (tid == 0) atomicAdd(g_sum, s_se);
}

// h2 = elu(g_h2/(S+1e-16)+bias); out = relu(gru(h2,out)); d_out = out.lin2_w + lin2_b
__global__ void k_final(const float* g_h2, const float* g_sum, const float* mc_bias,
                        const float* outv, const float* mg_wih, const float* mg_whh,
                        const float* mg_bih, const float* mg_bhh,
                        const float* lin2_w, const float* lin2_b, float* out){
  __shared__ float s_h2[64], s_o[64], s_red[64];
  int c = threadIdx.x;
  float h2 = g_h2[c] / (*g_sum + 1e-16f) + mc_bias[c];
  h2 = h2 > 0.f ? h2 : (expf(h2) - 1.f);
  s_h2[c] = h2; s_o[c] = outv[c];
  __syncthreads();
  float gi[3], gh[3];
  for (int g = 0; g < 3; ++g){
    int k = g * 64 + c;
    float a = mg_bih[k], b = mg_bhh[k];
    for (int j = 0; j < 64; ++j){
      a += mg_wih[k * 64 + j] * s_h2[j];
      b += mg_whh[k * 64 + j] * s_o[j];
    }
    gi[g] = a; gh[g] = b;
  }
  float r = sigm(gi[0] + gh[0]);
  float z = sigm(gi[1] + gh[1]);
  float nn = tanhf(gi[2] + r * gh[2]);
  float o2 = (1.f - z) * nn + z * s_o[c];
  o2 = o2 > 0.f ? o2 : 0.f;
  s_red[c] = o2 * lin2_w[c];
  __syncthreads();
  if (c == 0){
    float s = lin2_b[0];
    for (int i = 0; i < 64; ++i) s += s_red[i];
    out[0] = s;
  }
}

// ---------------- host launcher ----------------

extern "C" void kernel_launch(void* const* d_in, const int* in_sizes, int n_in,
                              void* d_out, int out_size, void* d_ws, size_t ws_size,
                              hipStream_t stream)
{
  const long Nn = in_sizes[0];     // node_attr (N,1)
  const long E  = in_sizes[1];     // edge_attr (E,1)
  const float* node_attr = (const float*)d_in[0];
  const float* edge_attr = (const float*)d_in[1];
  const int*   eidx      = (const int*)d_in[2];
  const int* src = eidx;
  const int* dst = eidx + E;
  const float* lin1_w  = (const float*)d_in[3];
  const float* lin1_b  = (const float*)d_in[4];
  const float* gc1     = (const float*)d_in[5];
  const float* gc2     = (const float*)d_in[6];
  const float* att_l   = (const float*)d_in[7];
  const float* att_r   = (const float*)d_in[8];
  const float* gc_bias = (const float*)d_in[9];
  const float* gru_wih = (const float*)d_in[10];
  const float* gru_whh = (const float*)d_in[11];
  const float* gru_bih = (const float*)d_in[12];
  const float* gru_bhh = (const float*)d_in[13];
  const float* mc_lin  = (const float*)d_in[14];
  const float* att_src = (const float*)d_in[15];
  const float* att_dst = (const float*)d_in[16];
  const float* mc_bias = (const float*)d_in[17];
  const float* mg_wih  = (const float*)d_in[18];
  const float* mg_whh  = (const float*)d_in[19];
  const float* mg_bih  = (const float*)d_in[20];
  const float* mg_bhh  = (const float*)d_in[21];
  const float* lin2_w  = (const float*)d_in[22];
  const float* lin2_b  = (const float*)d_in[23];

  char* p = (char*)d_ws;
  auto carve = [&](size_t bytes)->char*{
    char* q = p; p += (bytes + 255) & ~(size_t)255; return q;
  };
  float*    x      = (float*)   carve((size_t)Nn * 64 * 4);
  _Float16* xh     = (_Float16*)carve((size_t)Nn * 64 * 2);
  float*    dstdot = (float*)   carve((size_t)Nn * 4);
  float*    amax   = (float*)   carve((size_t)Nn * 4);
  float*    ssum   = (float*)   carve((size_t)Nn * 4);
  float*    alpha  = (float*)   carve((size_t)E * 4);
  float*    hacc   = (float*)   carve((size_t)Nn * 64 * 4);
  float*    x2     = (float*)   carve((size_t)Nn * 64 * 4);
  _Float16* x2h    = (_Float16*)carve((size_t)Nn * 64 * 2);
  float*    araw   = (float*)   carve((size_t)Nn * 4);
  _Float16* w1h    = (_Float16*)carve(64 * 65 * 2);
  _Float16* w2h    = (_Float16*)carve(64 * 64 * 2);
  _Float16* wihh   = (_Float16*)carve(192 * 64 * 2);
  _Float16* whhh   = (_Float16*)carve(192 * 64 * 2);
  _Float16* mclh   = (_Float16*)carve(64 * 64 * 2);
  float*    gpool  = (float*)   carve(64 * 4);
  float*    outv   = (float*)   carve(64 * 4);
  float*    g_h2   = (float*)   carve(64 * 4);
  float*    c0     = (float*)   carve(4);
  float*    g_sum  = (float*)   carve(4);
  float*    g_amax = (float*)   carve(4);
  float*    xs     = hacc;   // hacc dead after k_node_gru; reuse for readout xs

  const long etiles = (E + 15) / 16;
  const long ntiles = (Nn + 15) / 16;
  const int eblk  = (int)((etiles + (long)WPB * TPW - 1) / ((long)WPB * TPW));
  const int rblk  = (int)((ntiles + (long)WPB * TPW - 1) / ((long)WPB * TPW));
  const int gblk  = (int)((ntiles + WPBG - 1) / WPBG);

  k_init<<<(int)((Nn * 64 + 255) / 256), 256, 0, stream>>>(
      hacc, amax, ssum, gpool, g_h2, g_sum, g_amax, Nn);
  k_conv_w<<<(192 * 64 + 255) / 256, 256, 0, stream>>>(
      gc1, gc2, gru_wih, gru_whh, mc_lin, w1h, w2h, wihh, whhh, mclh);
  k_node_prep<<<(int)((Nn + WPB - 1) / WPB), 256, 0, stream>>>(
      node_attr, lin1_w, lin1_b, att_r, x, xh, dstdot, Nn);

  k_edge1<<<eblk, 256, 0, stream>>>(
      xh, edge_attr, src, dst, w1h, gc1, att_l, dstdot, alpha, amax, E);
  k_edge2<<<(int)((E + 255) / 256), 256, 0, stream>>>(dst, alpha, amax, ssum, E);
  k_edge3<<<eblk, 256, 0, stream>>>(
      xh, edge_attr, src, dst, w1h, gc1, w2h, alpha, ssum, hacc, E);

  k_node_gru<<<gblk, 128, 0, stream>>>(
      hacc, gc_bias, x, xh, wihh, whhh, gru_bih, gru_bhh, x2, x2h, gpool, Nn);

  k_pool_fin<<<1, 64, 0, stream>>>(gpool, mc_lin, att_dst, outv, c0);
  k_readout_xs<<<rblk, 256, 0, stream>>>(
      x2h, mclh, att_src, c0, xs, araw, g_amax, Nn);
  k_readout_sum<<<(int)((Nn + 255) / 256), 256, 0, stream>>>(
      xs, araw, g_amax, g_sum, g_h2, Nn);
  k_final<<<1, 64, 0, stream>>>(
      g_h2, g_sum, mc_bias, outv, mg_wih, mg_whh, mg_bih, mg_bhh,
      lin2_w, lin2_b, (float*)d_out);
  (void)n_in; (void)out_size; (void)ws_size;
}
